// TNDconf_54769422958933
// MI455X (gfx1250) — matrix operand order repaired and checked by reading, not compiled, not run on hardware
//
#include <hip/hip_runtime.h>
#include <hip/hip_bf16.h>
#include <math.h>

#define NN 30000
#define TT 6
#define EE 960000
#define XDIM 128
#define HD 256
#define ZDIM 128
#define GK 160        // GRU input K padded 129 -> 160 (multiple of 32)
#define G3 768        // 3*HD

// ---------------- WMMA GEMM (bf16 inputs, f32 accumulate) ----------------
typedef __attribute__((ext_vector_type(16))) __bf16 v16bf;
typedef __attribute__((ext_vector_type(8)))  float  v8f;
typedef __attribute__((ext_vector_type(4)))  int    v4i;

union FragAB { v16bf v; uint4 q[2]; };

__device__ __forceinline__ unsigned short f2bs(float f) {
  unsigned u = __float_as_uint(f);
  u += 0x7FFFu + ((u >> 16) & 1u);   // round-to-nearest-even to bf16
  return (unsigned short)(u >> 16);
}

#define BM 128
#define BN 64
#define BK 32
#define LDSTR 40   // padded LDS row stride in bf16 elements (80B, 16B aligned)

#define AS1 __attribute__((address_space(1)))
#define AS3 __attribute__((address_space(3)))

#if defined(__gfx1250__) && __has_builtin(__builtin_amdgcn_global_load_async_to_lds_b128)
#define HAVE_ASYNC 1
#else
#define HAVE_ASYNC 0
#endif

__device__ __forceinline__ void wait_async0() {
#if __has_builtin(__builtin_amdgcn_s_wait_asynccnt)
  __builtin_amdgcn_s_wait_asynccnt(0);
#else
  asm volatile("s_wait_asynccnt 0x0" ::: "memory");
#endif
}

// C[Nrows x M] = act(A[Nrows x K] @ B[K x M] + bias)
// A row-major f32; B supplied pre-transposed as bf16: Bt[n*K + k] = bf16(B[k][n]).
// K must be a multiple of 32. act: 0=none, 1=relu.
__global__ __launch_bounds__(256)
void wmma_gemm(const float* __restrict__ A, const unsigned short* __restrict__ Bt,
               const float* __restrict__ bias, float* __restrict__ C,
               int Nrows, int K, int M, int act)
{
  __shared__ unsigned short As[BM * LDSTR];
  __shared__ unsigned short Bs[BN * LDSTR];

  const int tid  = threadIdx.x;
  const int lane = tid & 31;
  const int wave = tid >> 5;          // 8 waves (wave32)
  const int wm = (wave & 3) * 32;     // 4 wave-rows, 32 rows each
  const int wn = (wave >> 2) * 32;    // 2 wave-cols, 32 cols each
  const int mb = blockIdx.x * BM;
  const int nb = blockIdx.y * BN;
  const int half = lane >> 4;
  const int l16  = lane & 15;

  v8f acc[2][2] = {};

  const int arow = tid >> 1;          // 0..127
  const int acol = (tid & 1) * 16;    // 0,16
  const int brow = wave * 8 + (lane >> 2);   // 0..63
  const int bseg = (lane & 3) * 8;           // 0,8,16,24 (bf16 elements)

  for (int kt = 0; kt < K; kt += BK) {
    // ---- stage A tile (128 rows x 32 K): f32 -> bf16 in-kernel ----
    {
      const int gr = mb + arow;
      float tmp[16];
      if (gr < Nrows) {
        const float* p = A + (size_t)gr * K + kt + acol;
        #pragma unroll
        for (int j = 0; j < 16; ++j) tmp[j] = p[j];
      } else {
        #pragma unroll
        for (int j = 0; j < 16; ++j) tmp[j] = 0.0f;
      }
      unsigned short* d = &As[arow * LDSTR + acol];
      #pragma unroll
      for (int j = 0; j < 16; ++j) d[j] = f2bs(tmp[j]);
    }
    // ---- stage B tile (64 cols x 32 K): one async 16B copy per lane ----
    {
      const int gn = nb + brow;
      unsigned short* ld = &Bs[brow * LDSTR + bseg];
      if (gn < M) {
        const unsigned short* gp = Bt + (size_t)gn * K + kt + bseg;
#if HAVE_ASYNC
        AS1 v4i* gsrc = (AS1 v4i*)gp;   // addrspacecast + bitcast (drops const)
        AS3 v4i* ldst = (AS3 v4i*)ld;
        __builtin_amdgcn_global_load_async_to_lds_b128(gsrc, ldst, 0, 0);
#else
        *(uint4*)ld = *(const uint4*)gp;
#endif
      } else {
        uint4 z = {};
        *(uint4*)ld = z;
      }
    }
#if HAVE_ASYNC
    wait_async0();
#endif
    __syncthreads();

    // ---- 2x2 WMMA per wave (32x32 output), fragments per ISA layouts ----
    #pragma unroll
    for (int mi = 0; mi < 2; ++mi) {
      FragAB fa;
      const unsigned short* pa = &As[(wm + mi * 16 + l16) * LDSTR];
      fa.q[0] = *(const uint4*)(pa + half * 8);        // K = half*8 + [0,8)
      fa.q[1] = *(const uint4*)(pa + 16 + half * 8);   // K = 16+half*8 + [0,8)
      #pragma unroll
      for (int ni = 0; ni < 2; ++ni) {
        FragAB fb;
        const unsigned short* pb = &Bs[(wn + ni * 16 + l16) * LDSTR];
        fb.q[0] = *(const uint4*)(pb + half * 8);
        fb.q[1] = *(const uint4*)(pb + 16 + half * 8);
        acc[mi][ni] = __builtin_amdgcn_wmma_f32_16x16x32_bf16(
            false, fa.v, false, fb.v, (short)0, acc[mi][ni], false, false);
      }
    }
    __syncthreads();
  }

  // ---- epilogue: C/D layout (vgpr r -> row +half*8+r, col l16) ----
  #pragma unroll
  for (int mi = 0; mi < 2; ++mi) {
    const int rbase = mb + wm + mi * 16 + half * 8;
    #pragma unroll
    for (int ni = 0; ni < 2; ++ni) {
      const int col = nb + wn + ni * 16 + l16;
      if (col < M) {
        const float bb = bias ? bias[col] : 0.0f;
        #pragma unroll
        for (int r = 0; r < 8; ++r) {
          const int row = rbase + r;
          if (row < Nrows) {
            float v = acc[mi][ni][r] + bb;
            if (act) v = fmaxf(v, 0.0f);
            C[(size_t)row * M + col] = v;
          }
        }
      }
    }
  }
}

// ---------------- weight pre-conversion to transposed bf16 ----------------
// Bt[M][K] = bf16(W[K][M])  (transpose; K multiple of 32)
__global__ void convT_k(const float* __restrict__ in, unsigned short* __restrict__ out,
                        int K, int M) {
  int i = blockIdx.x * blockDim.x + threadIdx.x;
  if (i >= M * K) return;
  int n = i / K, k = i - n * K;
  out[i] = f2bs(in[(size_t)k * M + n]);
}
// Bt[M][KP] = bf16(W[M][Ksrc]) zero-padded along K (no transpose: W already [M x Ksrc])
__global__ void convPad_k(const float* __restrict__ in, unsigned short* __restrict__ out,
                          int M, int Ksrc, int KP) {
  int i = blockIdx.x * blockDim.x + threadIdx.x;
  if (i >= M * KP) return;
  int n = i / KP, k = i - n * KP;
  out[i] = (k < Ksrc) ? f2bs(in[(size_t)n * Ksrc + k]) : (unsigned short)0;
}

// ---------------- SpMM: agg[dst] += w * sup[src]  (wave per edge) ----------------
__global__ __launch_bounds__(256)
void spmm_edges(const float* __restrict__ w, const int* __restrict__ src,
                const int* __restrict__ dst, const float* __restrict__ sup,
                float* __restrict__ agg, int E)
{
  const int lane = threadIdx.x & 31;
  const int gw = blockIdx.x * (blockDim.x >> 5) + (threadIdx.x >> 5);
  const int nw = gridDim.x * (blockDim.x >> 5);
  for (int e = gw; e < E; e += nw) {
    const int s = src[e], d = dst[e];
    const float wt = w[e];
    const float* ps = sup + (size_t)s * HD;
    float* pd = agg + (size_t)d * HD;
    #pragma unroll
    for (int j = 0; j < 8; ++j) {
      const int col = lane + j * 32;
      __hip_atomic_fetch_add(pd + col, wt * ps[col],
                             __ATOMIC_RELAXED, __HIP_MEMORY_SCOPE_AGENT);
    }
  }
}

// ---------------- elementwise helpers ----------------
__global__ void fill_zero_k(float* p, int n) {
  int i = blockIdx.x * blockDim.x + threadIdx.x;
  if (i < n) p[i] = 0.0f;
}
__global__ void copy_k(const float* a, float* b, int n) {
  int i = blockIdx.x * blockDim.x + threadIdx.x;
  if (i < n) b[i] = a[i];
}
__global__ void concat_hr_k(const float* __restrict__ h, const float* __restrict__ rep,
                            float* __restrict__ cat, int n) {
  int i = blockIdx.x * blockDim.x + threadIdx.x;
  if (i >= n) return;
  int r = i >> 9, j = i & 511;
  cat[i] = (j < HD) ? h[(size_t)r * HD + j] : rep[(size_t)r * HD + (j - HD)];
}
__global__ void build_gin_k(const float* __restrict__ z, const int* __restrict__ c,
                            float* __restrict__ gin, int n) {
  int i = blockIdx.x * blockDim.x + threadIdx.x;
  if (i >= n) return;
  int r = i / GK, j = i - r * GK;
  float v = 0.0f;
  if (j < ZDIM) v = z[(size_t)r * ZDIM + j];
  else if (j == ZDIM) v = (float)c[r];
  gin[i] = v;
}
__global__ void bias_relu_hd_k(const float* __restrict__ agg, const float* __restrict__ b,
                               float* __restrict__ rep, int n) {
  int i = blockIdx.x * blockDim.x + threadIdx.x;
  if (i >= n) return;
  float v = agg[i] + b[i & (HD - 1)];
  rep[i] = v > 0.0f ? v : 0.0f;
}
__global__ void gru_gates_k(const float* __restrict__ gi, const float* __restrict__ gh,
                            const float* __restrict__ hold, float* __restrict__ hnew, int n) {
  int i = blockIdx.x * blockDim.x + threadIdx.x;
  if (i >= n) return;
  int r = i >> 8, j = i & 255;
  const float* pi = gi + (size_t)r * G3;
  const float* ph = gh + (size_t)r * G3;
  float rr = 1.0f / (1.0f + expf(-(pi[j] + ph[j])));
  float uu = 1.0f / (1.0f + expf(-(pi[HD + j] + ph[HD + j])));
  float nn = tanhf(pi[2 * HD + j] + rr * ph[2 * HD + j]);
  hnew[i] = (1.0f - uu) * nn + uu * hold[i];
}
__global__ __launch_bounds__(256)
void dot_head_k(const float* __restrict__ y, const float* __restrict__ wv,
                const float* __restrict__ b, float* __restrict__ out) {
  const int lane = threadIdx.x & 31;
  const int gw = blockIdx.x * (blockDim.x >> 5) + (threadIdx.x >> 5);
  const int nw = gridDim.x * (blockDim.x >> 5);
  for (int n = gw; n < NN; n += nw) {
    float s = 0.0f;
    #pragma unroll
    for (int j = 0; j < 4; ++j) {
      const int k = lane + j * 32;
      s += y[(size_t)n * ZDIM + k] * wv[k];
    }
    #pragma unroll
    for (int m = 16; m; m >>= 1) s += __shfl_xor(s, m, 32);
    if (lane == 0) out[n] = s + b[0];
  }
}
__global__ void bn_sigmoid_k(float* __restrict__ hbn, const float* __restrict__ g,
                             const float* __restrict__ be, const float* __restrict__ mu,
                             const float* __restrict__ va, int n) {
  int i = blockIdx.x * blockDim.x + threadIdx.x;
  if (i >= n) return;
  int j = i % 100;
  float v = hbn[i];
  v = (v - mu[j]) * rsqrtf(va[j] + 1e-5f) * g[j] + be[j];
  hbn[i] = 1.0f / (1.0f + expf(-v));
}
__global__ void fc2_softmax_k(const float* __restrict__ hbn, const float* __restrict__ W,
                              const float* __restrict__ b, float* __restrict__ ps) {
  int n = blockIdx.x * blockDim.x + threadIdx.x;
  if (n >= NN) return;
  const float* p = hbn + (size_t)n * 100;
  float a0 = b[0], a1 = b[1];
  for (int k = 0; k < 100; ++k) {
    float v = p[k];
    a0 += v * W[2 * k];
    a1 += v * W[2 * k + 1];
  }
  float m = fmaxf(a0, a1);
  float e0 = expf(a0 - m), e1 = expf(a1 - m);
  float s = e0 + e1;
  ps[2 * n] = e0 / s;
  ps[2 * n + 1] = e1 / s;
}

// ---------------- host side ----------------
static inline int cdiv(int a, int b) { return (a + b - 1) / b; }

static void launch_gemm(const float* A, const unsigned short* Bt, const float* bias,
                        float* C, int Nrows, int K, int M, int act, hipStream_t s) {
  dim3 g(cdiv(Nrows, BM), cdiv(M, BN));
  wmma_gemm<<<g, dim3(256), 0, s>>>(A, Bt, bias, C, Nrows, K, M, act);
}

#define EW(kern, n, ...) kern<<<cdiv((n), 256), 256, 0, stream>>>(__VA_ARGS__)

extern "C" void kernel_launch(void* const* d_in, const int* in_sizes, int n_in,
                              void* d_out, int out_size, void* d_ws, size_t ws_size,
                              hipStream_t stream) {
  (void)in_sizes; (void)n_in; (void)out_size; (void)ws_size;
  const float* X      = (const float*)d_in[0];
  const int*   C      = (const int*)  d_in[1];
  const float* w      = (const float*)d_in[2];
  const int*   src    = (const int*)  d_in[3];
  const int*   dst    = (const int*)  d_in[4];
  const float* W_phi  = (const float*)d_in[5];
  const float* b_phi  = (const float*)d_in[6];
  const float* W_gc   = (const float*)d_in[7];
  const float* b_gc   = (const float*)d_in[8];
  const float* W_fuse = (const float*)d_in[9];
  const float* b_fuse = (const float*)d_in[10];
  const float* W_ih   = (const float*)d_in[11];
  const float* W_hh   = (const float*)d_in[12];
  const float* b_ih   = (const float*)d_in[13];
  const float* b_hh   = (const float*)d_in[14];
  const float* W_00   = (const float*)d_in[15];
  const float* b_00   = (const float*)d_in[16];
  const float* W_10   = (const float*)d_in[17];
  const float* b_10   = (const float*)d_in[18];
  const float* W_01   = (const float*)d_in[19];
  const float* b_01   = (const float*)d_in[20];
  const float* W_11   = (const float*)d_in[21];
  const float* b_11   = (const float*)d_in[22];
  const float* W_fc1  = (const float*)d_in[23];
  const float* b_fc1  = (const float*)d_in[24];
  const float* bn_g   = (const float*)d_in[25];
  const float* bn_b   = (const float*)d_in[26];
  const float* bn_m   = (const float*)d_in[27];
  const float* bn_v   = (const float*)d_in[28];
  const float* W_fc2  = (const float*)d_in[29];
  const float* b_fc2  = (const float*)d_in[30];
  const float* h0     = (const float*)d_in[31];

  float* out = (float*)d_out;
  const size_t oY1 = 0;
  const size_t oY0 = (size_t)TT * NN;
  const size_t oZ  = 2ull * TT * NN;
  const size_t oPS = oZ + (size_t)TT * NN * ZDIM;
  const size_t oH  = oPS + (size_t)TT * NN * 2;

  // ---- workspace: bf16 transposed weights first, then f32 buffers ----
  unsigned short* wsu = (unsigned short*)d_ws;
  size_t uoff = 0;
  unsigned short* TBphi = wsu + uoff; uoff += (size_t)HD * XDIM;      // [256][128]
  unsigned short* TBgc0 = wsu + uoff; uoff += (size_t)HD * HD;        // [256][256]
  unsigned short* TBgc1 = wsu + uoff; uoff += (size_t)HD * HD;
  unsigned short* TBfu  = wsu + uoff; uoff += (size_t)ZDIM * 2 * HD;  // [128][512]
  unsigned short* TBih  = wsu + uoff; uoff += (size_t)G3 * GK;        // [768][160]
  unsigned short* TBhh  = wsu + uoff; uoff += (size_t)G3 * HD;        // [768][256]
  unsigned short* TB00  = wsu + uoff; uoff += (size_t)ZDIM * ZDIM;    // [128][128]
  unsigned short* TB10  = wsu + uoff; uoff += (size_t)ZDIM * ZDIM;
  unsigned short* TBfc1 = wsu + uoff; uoff += (size_t)100 * ZDIM;     // [100][128]
  uoff = (uoff + 7) & ~(size_t)7;   // 16B align for the float region

  float* ws = (float*)(wsu + uoff);
  size_t off = 0;
  float* REP = ws + off; off += (size_t)NN * HD;
  float* SUP = ws + off; off += (size_t)NN * HD;
  float* AGG = ws + off; off += (size_t)NN * HD;
  float* CAT = ws + off; off += (size_t)NN * 2 * HD;  // fuse-in; reused: gru-in, heads
  float* GI  = ws + off; off += (size_t)NN * G3;
  float* GH  = ws + off; off += (size_t)NN * G3;
  float* HA  = ws + off; off += (size_t)NN * HD;
  float* HB  = ws + off; off += (size_t)NN * HD;

  // ---- per-launch weight conversion (tiny, L2-resident thereafter) ----
  EW(convT_k,  HD * XDIM,      W_phi,  TBphi, XDIM,    HD);
  EW(convT_k,  HD * HD,        W_gc,   TBgc0, HD,      HD);
  EW(convT_k,  HD * HD,        W_gc + (size_t)HD * HD, TBgc1, HD, HD);
  EW(convT_k,  ZDIM * 2 * HD,  W_fuse, TBfu,  2 * HD,  ZDIM);
  EW(convPad_k, G3 * GK,       W_ih,   TBih,  G3, ZDIM + 1, GK);   // already [M x K]
  EW(convPad_k, G3 * HD,       W_hh,   TBhh,  G3, HD,       HD);
  EW(convT_k,  ZDIM * ZDIM,    W_00,   TB00,  ZDIM, ZDIM);
  EW(convT_k,  ZDIM * ZDIM,    W_10,   TB10,  ZDIM, ZDIM);
  EW(convT_k,  100 * ZDIM,     W_fc1,  TBfc1, ZDIM, 100);
  EW(copy_k,   NN * HD, h0, HA, NN * HD);

  float* hcur = HA;
  float* hnext = HB;

  for (int t = 0; t < TT; ++t) {
    const float* Xt = X + (size_t)t * NN * XDIM;
    float* Zt = out + oZ + (size_t)t * NN * ZDIM;

    // phi = relu(X @ W_phi + b_phi)
    launch_gemm(Xt, TBphi, b_phi, REP, NN, XDIM, HD, 1, stream);

    // 2x GCN layer: sup = rep @ W_gc[l]; agg = scatter-add; rep = relu(agg + b)
    for (int l = 0; l < 2; ++l) {
      launch_gemm(REP, l ? TBgc1 : TBgc0, nullptr, SUP, NN, HD, HD, 0, stream);
      EW(fill_zero_k, NN * HD, AGG, NN * HD);
      spmm_edges<<<4096, 256, 0, stream>>>(w + (size_t)t * EE, src + (size_t)t * EE,
                                           dst + (size_t)t * EE, SUP, AGG, EE);
      EW(bias_relu_hd_k, NN * HD, AGG, b_gc + (size_t)l * HD, REP, NN * HD);
    }

    // z = relu([h, rep] @ W_fuse + b_fuse)  -> straight into output Z[t]
    EW(concat_hr_k, NN * 2 * HD, hcur, REP, CAT, NN * 2 * HD);
    launch_gemm(CAT, TBfu, b_fuse, Zt, NN, 2 * HD, ZDIM, 1, stream);

    // GRU: gi = [z,c,pad] @ W_ih^T + b_ih ; gh = h @ W_hh^T + b_hh ; gates
    EW(build_gin_k, NN * GK, Zt, C + (size_t)t * NN, CAT, NN * GK);
    launch_gemm(CAT, TBih, b_ih, GI, NN, GK, G3, 0, stream);
    launch_gemm(hcur, TBhh, b_hh, GH, NN, HD, G3, 0, stream);
    EW(gru_gates_k, NN * HD, GI, GH, hcur, hnext, NN * HD);

    // heads (CAT reused as scratch)
    launch_gemm(Zt, TB00, b_00, CAT, NN, ZDIM, ZDIM, 1, stream);
    dot_head_k<<<512, 256, 0, stream>>>(CAT, W_01, b_01, out + oY0 + (size_t)t * NN);
    launch_gemm(Zt, TB10, b_10, CAT, NN, ZDIM, ZDIM, 1, stream);
    dot_head_k<<<512, 256, 0, stream>>>(CAT, W_11, b_11, out + oY1 + (size_t)t * NN);

    launch_gemm(Zt, TBfc1, b_fc1, CAT, NN, ZDIM, 100, 0, stream);
    EW(bn_sigmoid_k, NN * 100, CAT, bn_g, bn_b, bn_m, bn_v, NN * 100);
    fc2_softmax_k<<<cdiv(NN, 256), 256, 0, stream>>>(CAT, W_fc2, b_fc2,
                                                     out + oPS + (size_t)t * NN * 2);

    float* tmp = hcur; hcur = hnext; hnext = tmp;
  }

  EW(copy_k, NN * HD, hcur, out + oH, NN * HD);
}
